// HunyuanTopKGate_78469052498380
// MI455X (gfx1250) — compile-verified, matrix operand order checked by reading, not run on hardware
//
#include <hip/hip_runtime.h>

typedef __bf16 bf16_t;
typedef __attribute__((ext_vector_type(16))) __bf16 v16bf;
typedef __attribute__((ext_vector_type(8)))  float  v8f;
typedef __attribute__((ext_vector_type(4)))  float  v4f;

#define TOKENS 16384
#define HID    4096
#define NEXP   64
#define KSPLIT 4
#define KSEG   (HID / KSPLIT)   // 1024
#define KSTEP  32

// Build a 16x32 (A) / 32x16 (B) bf16 WMMA fragment for this lane from fp32 rows.
// CDNA5 16-bit operand layout: lanes 0-15 hold K {0..7, 16..23}, lanes 16-31
// hold K {8..15, 24..31} -> per-lane base offsets half*8 and 16+half*8.
__device__ __forceinline__ v16bf load_frag(const float* __restrict__ rowp, int half) {
  const float* p = rowp + half * 8;
  v4f a0 = *(const v4f*)(p + 0);
  v4f a1 = *(const v4f*)(p + 4);
  v4f a2 = *(const v4f*)(p + 16);
  v4f a3 = *(const v4f*)(p + 20);
  v16bf f;
  f[0]  = (bf16_t)a0.x;  f[1]  = (bf16_t)a0.y;  f[2]  = (bf16_t)a0.z;  f[3]  = (bf16_t)a0.w;
  f[4]  = (bf16_t)a1.x;  f[5]  = (bf16_t)a1.y;  f[6]  = (bf16_t)a1.z;  f[7]  = (bf16_t)a1.w;
  f[8]  = (bf16_t)a2.x;  f[9]  = (bf16_t)a2.y;  f[10] = (bf16_t)a2.z;  f[11] = (bf16_t)a2.w;
  f[12] = (bf16_t)a3.x;  f[13] = (bf16_t)a3.y;  f[14] = (bf16_t)a3.z;  f[15] = (bf16_t)a3.w;
  return f;
}

#define WMMA_BF16(A, B, C) \
  __builtin_amdgcn_wmma_f32_16x16x32_bf16(false, (A), false, (B), (short)0, (C), false, false)

// partial = X (16384x4096) * W^T (4096x64) for one K segment; bf16 WMMA, f32 acc.
// One wave = one 16-row M tile x all 64 expert columns x one K segment.
// Software-pipelined one deep: fragments for k+1 are loaded/converted into fresh
// registers before the WMMAs for k issue -> no WMMA WAR-hazard v_nops, and the
// WMMAs overlap the next iteration's global loads.
__global__ __launch_bounds__(256) void gate_gemm_kernel(const float* __restrict__ X,
                                                        const float* __restrict__ W,
                                                        float* __restrict__ partial) {
  const int lane = threadIdx.x & 31;
  const int wave = threadIdx.x >> 5;
  const int half = lane >> 4;
  const int lrow = lane & 15;

  const int wid   = blockIdx.x * 8 + wave;     // 4096 waves total
  const int kseg  = wid & (KSPLIT - 1);
  const int tile  = wid >> 2;                  // 1024 M-tiles
  const int r0    = tile * 16;
  const int kbase = kseg * KSEG;

  const float* arow  = X + (size_t)(r0 + lrow) * HID + kbase;
  const float* brow0 = W + (size_t)( 0 + lrow) * HID + kbase;
  const float* brow1 = W + (size_t)(16 + lrow) * HID + kbase;
  const float* brow2 = W + (size_t)(32 + lrow) * HID + kbase;
  const float* brow3 = W + (size_t)(48 + lrow) * HID + kbase;

  v8f acc0 = {}, acc1 = {}, acc2 = {}, acc3 = {};

  // prologue: fragments for k0 = 0
  v16bf a  = load_frag(arow,  half);
  v16bf b0 = load_frag(brow0, half);
  v16bf b1 = load_frag(brow1, half);
  v16bf b2 = load_frag(brow2, half);
  v16bf b3 = load_frag(brow3, half);

  #pragma unroll 2
  for (int k0 = KSTEP; k0 < KSEG; k0 += KSTEP) {
    v16bf an  = load_frag(arow  + k0, half);
    v16bf bn0 = load_frag(brow0 + k0, half);
    v16bf bn1 = load_frag(brow1 + k0, half);
    v16bf bn2 = load_frag(brow2 + k0, half);
    v16bf bn3 = load_frag(brow3 + k0, half);

    acc0 = WMMA_BF16(a, b0, acc0);
    acc1 = WMMA_BF16(a, b1, acc1);
    acc2 = WMMA_BF16(a, b2, acc2);
    acc3 = WMMA_BF16(a, b3, acc3);

    a = an; b0 = bn0; b1 = bn1; b2 = bn2; b3 = bn3;
  }

  // epilogue: last K chunk
  acc0 = WMMA_BF16(a, b0, acc0);
  acc1 = WMMA_BF16(a, b1, acc1);
  acc2 = WMMA_BF16(a, b2, acc2);
  acc3 = WMMA_BF16(a, b3, acc3);

  // C/D layout: VGPR v -> row r0 + v + 8*half, column = lrow within each 16-col tile
  float* outp = partial + (size_t)kseg * TOKENS * NEXP;
  #pragma unroll
  for (int v = 0; v < 8; ++v) {
    const size_t row = (size_t)(r0 + v + 8 * half);
    outp[row * NEXP +  0 + lrow] = acc0[v];
    outp[row * NEXP + 16 + lrow] = acc1[v];
    outp[row * NEXP + 32 + lrow] = acc2[v];
    outp[row * NEXP + 48 + lrow] = acc3[v];
  }
}

// Per-token: reduce K-split partials, softmax over 64 experts, top-16 select.
__global__ __launch_bounds__(256) void gate_topk_kernel(const float* __restrict__ partial,
                                                        float* __restrict__ out) {
  const int t = blockIdx.x * blockDim.x + threadIdx.x;
  if (t >= TOKENS) return;

  float g[NEXP];
  {
    const float* p0 = partial + (size_t)t * NEXP;
    #pragma unroll
    for (int i = 0; i < NEXP; i += 4) {
      v4f s = *(const v4f*)(p0 + i);
      g[i] = s.x; g[i+1] = s.y; g[i+2] = s.z; g[i+3] = s.w;
    }
    #pragma unroll
    for (int p = 1; p < KSPLIT; ++p) {
      const float* pp = partial + (size_t)p * TOKENS * NEXP + (size_t)t * NEXP;
      #pragma unroll
      for (int i = 0; i < NEXP; i += 4) {
        v4f s = *(const v4f*)(pp + i);
        g[i] += s.x; g[i+1] += s.y; g[i+2] += s.z; g[i+3] += s.w;
      }
    }
  }

  // softmax (all static indexing -> g stays in VGPRs)
  float mx = g[0];
  #pragma unroll
  for (int i = 1; i < NEXP; ++i) mx = fmaxf(mx, g[i]);
  float sum = 0.0f;
  #pragma unroll
  for (int i = 0; i < NEXP; ++i) { g[i] = __expf(g[i] - mx); sum += g[i]; }
  const float inv = 1.0f / sum;
  #pragma unroll
  for (int i = 0; i < NEXP; ++i) g[i] *= inv;

  float* w_out = out;
  float* i_out = out + (size_t)TOKENS * 8;
  float* c_out = out + (size_t)2 * TOKENS * 8;
  float* a_out = out + (size_t)3 * TOKENS * 8;

  // top-16 via repeated masked argmax; strict '>' keeps lowest index on ties.
  // Outer loop rolled (compact code); inner argmax unrolled (g in registers).
  unsigned t_lo = 0u, t_hi = 0u;
  float s8 = 0.0f;
  #pragma unroll 1
  for (int k = 0; k < 16; ++k) {
    float best = -1.0f;
    int   bi   = 0;
    #pragma unroll
    for (int e = 0; e < NEXP; ++e) {
      const bool avail = (e < 32) ? ((t_lo & (1u << e)) == 0u)
                                  : ((t_hi & (1u << (e - 32))) == 0u);
      const bool better = avail && (g[e] > best);
      best = better ? g[e] : best;
      bi   = better ? e    : bi;
    }
    if (bi < 32) t_lo |= (1u << bi); else t_hi |= (1u << (bi - 32));
    if (k < 8) {
      w_out[(size_t)t * 8 + k] = best;       // unnormalized; renormalized below
      i_out[(size_t)t * 8 + k] = (float)bi;
      a_out[(size_t)t * 8 + k] = (float)bi;
      s8 += best;
    } else {
      c_out[(size_t)t * 8 + (k - 8)] = (float)bi;
    }
  }

  const float denom = fmaxf(s8, 1.1920929e-07f);  // FLT_EPSILON per reference
  // Same-wave same-address store->load is ordered (ISA 7.3): safe read-back.
  #pragma unroll
  for (int i = 0; i < 8; ++i) {
    w_out[(size_t)t * 8 + i] = w_out[(size_t)t * 8 + i] / denom;
  }
}

extern "C" void kernel_launch(void* const* d_in, const int* in_sizes, int n_in,
                              void* d_out, int out_size, void* d_ws, size_t ws_size,
                              hipStream_t stream) {
  const float* X = (const float*)d_in[0];   // (4,4096,4096) fp32 -> 16384 x 4096
  const float* W = (const float*)d_in[1];   // (64,4096) fp32
  float* out     = (float*)d_out;
  float* partial = (float*)d_ws;            // KSPLIT * 16384 * 64 fp32 = 16 MB

  // 1024 M-tiles * KSPLIT waves, 8 waves (256 thr) per block -> 512 blocks
  gate_gemm_kernel<<<dim3(512), dim3(256), 0, stream>>>(X, W, partial);
  gate_topk_kernel<<<dim3((TOKENS + 255) / 256), dim3(256), 0, stream>>>(partial, out);
}
